// get_model_7550552506749
// MI455X (gfx1250) — matrix operand order repaired and checked by reading, not compiled, not run on hardware
//
#include <hip/hip_runtime.h>
#include <hip/hip_bf16.h>
#include <cstdint>
#include <cstddef>

typedef __attribute__((ext_vector_type(16))) __bf16 v16bf;
typedef __attribute__((ext_vector_type(8)))  __bf16 v8bf;
typedef __attribute__((ext_vector_type(8)))  float  v8f;

#define B_     8
#define N_     4096
#define KN_    20
#define BN_    (B_ * N_)
#define BNK_   (BN_ * KN_)
#define EPS_   1e-5f
#define SLOPE_ 0.2f

enum { MODE_DENSE = 0, MODE_EDGE = 1, MODE_CAT3 = 2, MODE_CAT7 = 3 };

// ---------------------------------------------------------------- transpose
// x (B,6,N) -> xt (B,N,16) zero padded
__global__ void k_transpose(const float* __restrict__ x, float* __restrict__ xt) {
  int e = blockIdx.x * 256 + threadIdx.x;
  if (e >= BN_ * 16) return;
  int c = e & 15;
  int r = e >> 4;
  int b = r >> 12;
  int n = r & (N_ - 1);
  xt[e] = (c < 6) ? x[((size_t)b * 6 + c) * N_ + n] : 0.f;
}

// ---------------------------------------------------------------- fused kNN
// For each (b,n): scan all m, score = 2*<x_n,x_m> - |x_m|^2 (top-k invariant
// to the -|x_n|^2 constant), keep top-20 indices. LDS-tiled candidates.
template <int C, int LD, int TP>
__global__ void k_knn(const float* __restrict__ src, int* __restrict__ idxOut) {
  __shared__ float tile[TP][LD];
  const int b = blockIdx.y;
  const int n = blockIdx.x * 256 + threadIdx.x;
  const float* base = src + (size_t)b * N_ * LD;
  float f[C];
#pragma unroll
  for (int c = 0; c < C; ++c) f[c] = base[(size_t)n * LD + c];
  float vals[KN_];
  int   inds[KN_];
#pragma unroll
  for (int j = 0; j < KN_; ++j) { vals[j] = -3.0e38f; inds[j] = 0; }
  float vmin = -3.0e38f;
  int   minpos = 0;
  for (int m0 = 0; m0 < N_; m0 += TP) {
    __syncthreads();
    for (int e = threadIdx.x; e < TP * LD; e += 256)
      ((float*)tile)[e] = base[(size_t)m0 * LD + e];
    __syncthreads();
    for (int p = 0; p < TP; ++p) {
      float inner = 0.f, sq = 0.f;
#pragma unroll
      for (int c = 0; c < C; ++c) {
        float t = tile[p][c];
        inner = fmaf(f[c], t, inner);
        sq    = fmaf(t, t, sq);
      }
      float score = 2.f * inner - sq;
      if (score > vmin) {
        vals[minpos] = score;
        inds[minpos] = m0 + p;
        vmin = vals[0]; minpos = 0;
#pragma unroll
        for (int j = 1; j < KN_; ++j)
          if (vals[j] < vmin) { vmin = vals[j]; minpos = j; }
      }
    }
  }
  int* o = idxOut + ((size_t)b * N_ + n) * KN_;
#pragma unroll
  for (int j = 0; j < KN_; ++j) o[j] = inds[j];
}

// ---------------------------------------------------------------- WMMA GEMM
// Y[M,Nch] = A'[M,K] * W[Nch,K]^T ; A' fetched per MODE (fused gather/concat).
// Block: 256 thr = 8 waves; 128 rows x 64 cols per block; K chunked by 32.
// LDS tiles are laid out so each lane's WMMA fragment is two contiguous
// 16-byte runs -> ds_load_b128 instead of per-element b16 loads.
template <int MODE>
__global__ void k_gemm(const float* __restrict__ A, const float* __restrict__ W,
                       float* __restrict__ Y,
                       const float* __restrict__ src, const int* __restrict__ idx,
                       const float* __restrict__ x1, const float* __restrict__ x2,
                       const float* __restrict__ x3, const float* __restrict__ glob,
                       int M, int K, int Nch, int ldSrc, int Cin, int ldA) {
  __shared__ alignas(16) __bf16 lA[128][40];  // row stride 80 B (mult of 16)
  __shared__ alignas(16) __bf16 lBt[64][40];  // B stored transposed: [col][k]
  const int tid  = threadIdx.x;
  const int lane = tid & 31;
  const int wave = tid >> 5;
  const int half = lane >> 4;
  const int l16  = lane & 15;
  const int mBase = blockIdx.x * 128;
  const int nBase = blockIdx.y * 64;

  v8f accv[4];
#pragma unroll
  for (int t = 0; t < 4; ++t)
#pragma unroll
    for (int r = 0; r < 8; ++r) accv[t][r] = 0.f;

  const int nChunks = (K + 31) >> 5;
  for (int kc = 0; kc < nChunks; ++kc) {
    const int k0 = kc << 5;
    // --- stage A tile (fp32 -> bf16), mode-specific fetch -----------------
    for (int e = tid; e < 128 * 32; e += 256) {
      int r  = e >> 5, kk = e & 31;
      int gr = mBase + r;
      int c  = k0 + kk;
      float v = 0.f;
      if (gr < M && c < K) {
        if constexpr (MODE == MODE_DENSE) {
          v = A[(size_t)gr * ldA + c];
        } else if constexpr (MODE == MODE_EDGE) {
          unsigned ug  = (unsigned)gr;
          unsigned b   = ug / (N_ * KN_);
          unsigned rem = ug - b * (N_ * KN_);
          unsigned n   = rem / KN_;
          int i = idx[ug];
          if (c < Cin)
            v = src[((size_t)(b * N_ + (unsigned)i)) * ldSrc + c] -
                src[((size_t)(b * N_ + n)) * ldSrc + c];
          else
            v = src[((size_t)(b * N_ + n)) * ldSrc + (c - Cin)];
        } else if constexpr (MODE == MODE_CAT3) {
          int sel = c >> 6, cc = c & 63;
          const float* p = (sel == 0) ? x1 : (sel == 1) ? x2 : x3;
          v = p[(size_t)gr * 64 + cc];
        } else {  // MODE_CAT7: [glob(1024) | x1 | x2 | x3]
          if (c < 1024) {
            v = glob[(size_t)(gr >> 12) * 1024 + c];
          } else {
            int c2 = c - 1024, sel = c2 >> 6;
            const float* p = (sel == 0) ? x1 : (sel == 1) ? x2 : x3;
            v = p[(size_t)gr * 64 + (c2 & 63)];
          }
        }
      }
      lA[r][kk] = (__bf16)v;
    }
    // --- stage B tile transposed (weights row-major (Nch,K)) --------------
    for (int e = tid; e < 32 * 64; e += 256) {
      int kk = e & 31, col = e >> 5;
      int c  = k0 + kk;
      float v = (c < K) ? W[(size_t)(nBase + col) * K + c] : 0.f;
      lBt[col][kk] = (__bf16)v;
    }
    __syncthreads();
    // --- fragments per ISA 7.12.2 layouts, vectorized LDS reads -----------
    const int arow = wave * 16 + l16;
    v8bf alo = *(const v8bf*)&lA[arow][half * 8];        // K = half*8 .. +7
    v8bf ahi = *(const v8bf*)&lA[arow][16 + half * 8];   // K = 16+half*8 .. +7
    v16bf a = __builtin_shufflevector(alo, ahi,
        0, 1, 2, 3, 4, 5, 6, 7, 8, 9, 10, 11, 12, 13, 14, 15);
#pragma unroll
    for (int t = 0; t < 4; ++t) {
      const int col = t * 16 + l16;
      v8bf blo = *(const v8bf*)&lBt[col][half * 8];       // K = half*8 .. +7
      v8bf bhi = *(const v8bf*)&lBt[col][16 + half * 8];  // K = 16+half*8 .. +7
      v16bf bfr = __builtin_shufflevector(blo, bhi,
          0, 1, 2, 3, 4, 5, 6, 7, 8, 9, 10, 11, 12, 13, 14, 15);
      accv[t] = __builtin_amdgcn_wmma_f32_16x16x32_bf16(
          false, a, false, bfr, (short)0, accv[t], false, false);
    }
    __syncthreads();
  }
  // --- epilogue: fp32 pre-BN activations ----------------------------------
#pragma unroll
  for (int t = 0; t < 4; ++t) {
    const int col = nBase + t * 16 + l16;
#pragma unroll
    for (int r = 0; r < 8; ++r) {
      const int row = mBase + wave * 16 + half * 8 + r;
      if (row < M) Y[(size_t)row * Nch + col] = accv[t][r];
    }
  }
}

// ---------------------------------------------------------------- BN stats
__global__ void k_zero(float* p, int n) {
  int e = blockIdx.x * 256 + threadIdx.x;
  if (e < n) p[e] = 0.f;
}

__global__ void k_stats(const float* __restrict__ Y, long long M, int Nch,
                        float* __restrict__ stats) {
  __shared__ float ssum[1024];
  __shared__ float ssq[1024];
  for (int c = threadIdx.x; c < Nch; c += 256) { ssum[c] = 0.f; ssq[c] = 0.f; }
  __syncthreads();
  long long total  = M * (long long)Nch;
  long long stride = (long long)gridDim.x * 256;
  for (long long e = (long long)blockIdx.x * 256 + threadIdx.x; e < total; e += stride) {
    float v = Y[e];
    int   c = (int)e & (Nch - 1);  // Nch is always a power of two here
    atomicAdd(&ssum[c], v);
    atomicAdd(&ssq[c], v * v);
  }
  __syncthreads();
  for (int c = threadIdx.x; c < Nch; c += 256) {
    atomicAdd(&stats[c], ssum[c]);
    atomicAdd(&stats[Nch + c], ssq[c]);
  }
}

__global__ void k_bn_lrelu(float* __restrict__ Y, long long M, int Nch,
                           const float* __restrict__ stats,
                           const float* __restrict__ g, const float* __restrict__ bb) {
  long long total  = M * (long long)Nch;
  long long stride = (long long)gridDim.x * 256;
  float inv = 1.f / (float)M;
  for (long long e = (long long)blockIdx.x * 256 + threadIdx.x; e < total; e += stride) {
    int c = (int)e & (Nch - 1);
    float m   = stats[c] * inv;
    float var = stats[Nch + c] * inv - m * m;
    float s   = rsqrtf(var + EPS_) * g[c];
    float y   = (Y[e] - m) * s + bb[c];
    Y[e] = (y > 0.f) ? y : SLOPE_ * y;
  }
}

// BN + LeakyReLU + max over the k=20 neighbor axis, Y(BN*20,64) -> X(BN,64)
__global__ void k_bn_lrelu_maxk(const float* __restrict__ Y,
                                const float* __restrict__ stats,
                                const float* __restrict__ g,
                                const float* __restrict__ bb,
                                float* __restrict__ X) {
  int e = blockIdx.x * 256 + threadIdx.x;
  if (e >= BN_ * 64) return;
  int row = e >> 6, c = e & 63;
  const float inv = 1.f / (float)BNK_;
  float m   = stats[c] * inv;
  float var = stats[64 + c] * inv - m * m;
  float s   = rsqrtf(var + EPS_) * g[c];
  float b0  = bb[c];
  float best = -3.0e38f;
#pragma unroll
  for (int j = 0; j < KN_; ++j) {
    float y = (Y[((size_t)row * KN_ + j) * 64 + c] - m) * s + b0;
    y = (y > 0.f) ? y : SLOPE_ * y;
    best = fmaxf(best, y);
  }
  X[(size_t)row * 64 + c] = best;
}

// global max over N: H6 (B,N,1024) -> glob (B,1024); coalesced over channels
__global__ void k_maxn(const float* __restrict__ H6, float* __restrict__ glob) {
  int c = blockIdx.x * 256 + threadIdx.x;
  int b = blockIdx.y;
  float best = -3.0e38f;
  for (int n = 0; n < N_; ++n)
    best = fmaxf(best, H6[((size_t)(b * N_ + n)) * 1024 + c]);
  glob[b * 1024 + c] = best;
}

// classifier: out (B,17,N) = H8 (B,N,256) x Wcls(17,256)^T
__global__ void k_cls(const float* __restrict__ H8, const float* __restrict__ Wc,
                      float* __restrict__ out) {
  int e = blockIdx.x * 256 + threadIdx.x;
  if (e >= BN_ * 17) return;
  int o = e % 17;
  int r = e / 17;
  const float* h = H8 + (size_t)r * 256;
  const float* w = Wc + (size_t)o * 256;
  float s = 0.f;
  for (int c = 0; c < 256; ++c) s = fmaf(h[c], w[c], s);
  int b = r >> 12, n = r & (N_ - 1);
  out[((size_t)(b * 17 + o)) * N_ + n] = s;
}

// ================================================================= driver
extern "C" void kernel_launch(void* const* d_in, const int* in_sizes, int n_in,
                              void* d_out, int out_size, void* d_ws, size_t ws_size,
                              hipStream_t stream) {
  (void)in_sizes; (void)n_in; (void)out_size; (void)ws_size;
  const float* x = (const float*)d_in[0];
  const float *W[9], *G[9], *Bp[9];
  for (int j = 1; j <= 8; ++j) {
    W[j]  = (const float*)d_in[3 * j - 2];
    G[j]  = (const float*)d_in[3 * j - 1];
    Bp[j] = (const float*)d_in[3 * j];
  }
  const float* Wcls = (const float*)d_in[25];
  float* out = (float*)d_out;

  size_t off = 0;
  auto alloc = [&](size_t bytes) -> char* {
    char* p = (char*)d_ws + off;
    off += (bytes + 255) & ~(size_t)255;
    return p;
  };
  float* xt6  = (float*)alloc((size_t)BN_ * 16 * 4);
  float* X1   = (float*)alloc((size_t)BN_ * 64 * 4);
  float* X2   = (float*)alloc((size_t)BN_ * 64 * 4);
  float* X3   = (float*)alloc((size_t)BN_ * 64 * 4);
  int*   idx  = (int*)  alloc((size_t)BNK_ * 4);
  float* Y0   = (float*)alloc((size_t)BNK_ * 64 * 4);  // also reused as H6/H8
  float* Y1   = (float*)alloc((size_t)BNK_ * 64 * 4);  // also reused as H7
  float* glob = (float*)alloc((size_t)B_ * 1024 * 4);
  float* stats= (float*)alloc(2048 * 4);

  dim3 blk(256);
  const int gEdge = BNK_ / 128;  // 5120
  const int gRow  = BN_ / 128;   // 256

  auto run_stats = [&](const float* Y, long long M, int Nch) {
    k_zero<<<8, blk, 0, stream>>>(stats, 2048);
    k_stats<<<1024, blk, 0, stream>>>(Y, M, Nch, stats);
  };
  auto run_stats_bn = [&](float* Y, long long M, int Nch, const float* g, const float* b) {
    run_stats(Y, M, Nch);
    k_bn_lrelu<<<4096, blk, 0, stream>>>(Y, M, Nch, stats, g, b);
  };

  // ---- stage 0: transpose + first kNN (C=6) ----
  k_transpose<<<(BN_ * 16 + 255) / 256, blk, 0, stream>>>(x, xt6);
  k_knn<6, 16, 128><<<dim3(N_ / 256, B_), blk, 0, stream>>>(xt6, idx);

  // ---- layer 1: edge conv 12 -> 64 ----
  k_gemm<MODE_EDGE><<<dim3(gEdge, 1), blk, 0, stream>>>(
      nullptr, W[1], Y0, xt6, idx, nullptr, nullptr, nullptr, nullptr,
      BNK_, 12, 64, 16, 6, 0);
  run_stats_bn(Y0, BNK_, 64, G[1], Bp[1]);

  // ---- layer 2: 64 -> 64, then max over k -> X1 ----
  k_gemm<MODE_DENSE><<<dim3(gEdge, 1), blk, 0, stream>>>(
      Y0, W[2], Y1, nullptr, nullptr, nullptr, nullptr, nullptr, nullptr,
      BNK_, 64, 64, 0, 0, 64);
  run_stats(Y1, BNK_, 64);
  k_bn_lrelu_maxk<<<(BN_ * 64 + 255) / 256, blk, 0, stream>>>(Y1, stats, G[2], Bp[2], X1);

  // ---- kNN on X1, layers 3/4 ----
  k_knn<64, 64, 64><<<dim3(N_ / 256, B_), blk, 0, stream>>>(X1, idx);
  k_gemm<MODE_EDGE><<<dim3(gEdge, 1), blk, 0, stream>>>(
      nullptr, W[3], Y0, X1, idx, nullptr, nullptr, nullptr, nullptr,
      BNK_, 128, 64, 64, 64, 0);
  run_stats_bn(Y0, BNK_, 64, G[3], Bp[3]);
  k_gemm<MODE_DENSE><<<dim3(gEdge, 1), blk, 0, stream>>>(
      Y0, W[4], Y1, nullptr, nullptr, nullptr, nullptr, nullptr, nullptr,
      BNK_, 64, 64, 0, 0, 64);
  run_stats(Y1, BNK_, 64);
  k_bn_lrelu_maxk<<<(BN_ * 64 + 255) / 256, blk, 0, stream>>>(Y1, stats, G[4], Bp[4], X2);

  // ---- kNN on X2, layer 5 ----
  k_knn<64, 64, 64><<<dim3(N_ / 256, B_), blk, 0, stream>>>(X2, idx);
  k_gemm<MODE_EDGE><<<dim3(gEdge, 1), blk, 0, stream>>>(
      nullptr, W[5], Y0, X2, idx, nullptr, nullptr, nullptr, nullptr,
      BNK_, 128, 64, 64, 64, 0);
  run_stats(Y0, BNK_, 64);
  k_bn_lrelu_maxk<<<(BN_ * 64 + 255) / 256, blk, 0, stream>>>(Y0, stats, G[5], Bp[5], X3);

  // ---- layer 6: cat(x1,x2,x3) 192 -> 1024, global max ----
  float* H6 = Y0;
  k_gemm<MODE_CAT3><<<dim3(gRow, 1024 / 64), blk, 0, stream>>>(
      nullptr, W[6], H6, nullptr, nullptr, X1, X2, X3, nullptr,
      BN_, 192, 1024, 0, 0, 0);
  run_stats_bn(H6, BN_, 1024, G[6], Bp[6]);
  k_maxn<<<dim3(1024 / 256, B_), blk, 0, stream>>>(H6, glob);

  // ---- layer 7: cat(glob,x1,x2,x3) 1216 -> 512 ----
  float* H7 = Y1;
  k_gemm<MODE_CAT7><<<dim3(gRow, 512 / 64), blk, 0, stream>>>(
      nullptr, W[7], H7, nullptr, nullptr, X1, X2, X3, glob,
      BN_, 1216, 512, 0, 0, 0);
  run_stats_bn(H7, BN_, 512, G[7], Bp[7]);

  // ---- layer 8: 512 -> 256 ----
  float* H8 = Y0;
  k_gemm<MODE_DENSE><<<dim3(gRow, 256 / 64), blk, 0, stream>>>(
      H7, W[8], H8, nullptr, nullptr, nullptr, nullptr, nullptr, nullptr,
      BN_, 512, 256, 0, 0, 512);
  run_stats_bn(H8, BN_, 256, G[8], Bp[8]);

  // ---- classifier ----
  k_cls<<<(BN_ * 17 + 255) / 256, blk, 0, stream>>>(H8, Wcls, out);
}
